// AmplitudeQuantumNet_53369263620394
// MI455X (gfx1250) — compile-verified, matrix operand order checked by reading, not run on hardware
//
#include <hip/hip_runtime.h>
#include <hip/hip_bf16.h>
#include <math.h>

typedef __attribute__((ext_vector_type(16))) _Float16 v16h;
typedef __attribute__((ext_vector_type(8)))  float    v8f;

#define BATCH 1024

__device__ __forceinline__ v8f wmma_f16(v16h a, v16h b, v8f c) {
    return __builtin_amdgcn_wmma_f32_16x16x32_f16(
        /*neg_a=*/false, a, /*neg_b=*/false, b,
        /*c_mod=*/(short)0, c, /*reuse_a=*/false, /*reuse_b=*/false);
}

// ---------------------------------------------------------------------------
// K1: conv1 (1->32, 3x3, SAME) + BN1 + ReLU + 2x2 maxpool, fused.
// Writes channels-last, halo-padded f16 tensor [B][16][16][32]; the 1-pixel
// border is written as zero every call (no bounds checks needed in conv2,
// and the buffer is deterministically re-initialized for graph replay).
// ---------------------------------------------------------------------------
__global__ void k_conv1_pool_nhwc(const float* __restrict__ x,
                                  const float* __restrict__ w,
                                  const float* __restrict__ bias,
                                  const float* __restrict__ g,
                                  const float* __restrict__ be,
                                  const float* __restrict__ mu,
                                  const float* __restrict__ var,
                                  _Float16* __restrict__ out) {
    int idx = blockIdx.x * 256 + threadIdx.x;           // B*16*16*32
    const int total = BATCH * 16 * 16 * 32;
    if (idx >= total) return;
    int oc = idx & 31; int t = idx >> 5;
    int px = t & 15;   t >>= 4;
    int py = t & 15;   int b = t >> 4;

    if (py == 0 || py == 15 || px == 0 || px == 15) {   // halo
        out[idx] = (_Float16)0.0f;
        return;
    }
    const int ppy = py - 1, ppx = px - 1;               // pooled coords [0,14)

    float inv = g[oc] * rsqrtf(var[oc] + 1e-5f);
    float sh  = be[oc] - mu[oc] * inv;
    const float* wk = w + oc * 9;
    float m = 0.0f;  // ReLU floor; max(relu(v)) == max(0, max(v))
#pragma unroll
    for (int dy = 0; dy < 2; ++dy) {
#pragma unroll
        for (int dx = 0; dx < 2; ++dx) {
            int y = ppy * 2 + dy, xx = ppx * 2 + dx;
            float acc = bias[oc];
#pragma unroll
            for (int ky = 0; ky < 3; ++ky) {
#pragma unroll
                for (int kx = 0; kx < 3; ++kx) {
                    int iy = y + ky - 1, ix = xx + kx - 1;
                    if (iy >= 0 && iy < 28 && ix >= 0 && ix < 28)
                        acc = fmaf(x[(b * 28 + iy) * 28 + ix], wk[ky * 3 + kx], acc);
                }
            }
            m = fmaxf(m, acc * inv + sh);
        }
    }
    out[idx] = (_Float16)m;
}

// ---------------------------------------------------------------------------
// K2a: repack conv2_w OIHW f32 [64,32,3,3] -> f16 [tap r][oc][c]  (9,64,32)
// ---------------------------------------------------------------------------
__global__ void k_repack_w2(const float* __restrict__ w, _Float16* __restrict__ wt) {
    int i = blockIdx.x * 256 + threadIdx.x;             // 9*64*32 = 18432
    if (i >= 9 * 64 * 32) return;
    int c = i & 31;
    int n = (i >> 5) & 63;
    int r = i >> 11;
    wt[i] = (_Float16)w[(n * 32 + c) * 9 + r];
}

// ---------------------------------------------------------------------------
// K2b: generic f32 -> f16 conversion (fc weights)
// ---------------------------------------------------------------------------
__global__ void k_f32_to_f16(const float* __restrict__ in, _Float16* __restrict__ out, int n) {
    int i = blockIdx.x * 256 + threadIdx.x;
    if (i < n) out[i] = (_Float16)in[i];
}

// ---------------------------------------------------------------------------
// K3: conv2 as 9 shifted GEMMs via WMMA on padded NHWC input.
// M = B*196 (b,y,x), N = 64, K = 9 taps x 32 channels.
// One wave computes a 16x64 tile (4 accumulators share each A load).
// Epilogue fuses bias + BN2 + ReLU; writes f16 activations [B,64,14,14].
// ---------------------------------------------------------------------------
__global__ void k_conv2_wmma(const _Float16* __restrict__ a_in,  // [B,16,16,32]
                             const _Float16* __restrict__ wt,    // [9,64,32]
                             const float* __restrict__ bias,
                             const float* __restrict__ g,
                             const float* __restrict__ be,
                             const float* __restrict__ mu,
                             const float* __restrict__ var,
                             _Float16* __restrict__ act) {       // [B,64,14,14]
    const int lane  = threadIdx.x & 31;
    const int wv    = threadIdx.x >> 5;
    const int mbase = (blockIdx.x * 4 + wv) * 16;

    const int mrow = mbase + (lane & 15);
    const int b = mrow / 196;
    const int s = mrow - b * 196;
    const int y = s / 14, x = s - (s / 14) * 14;
    const int ko = (lane >> 4) * 16;
    const int nl = lane & 15;

    v8f acc0 = {}, acc1 = {}, acc2 = {}, acc3 = {};
#pragma unroll
    for (int r = 0; r < 9; ++r) {
        const int ky = r / 3, kx = r - (r / 3) * 3;
        v16h a  = *(const v16h*)(a_in + ((b * 16 + y + ky) * 16 + (x + kx)) * 32 + ko);
        v16h b0 = *(const v16h*)(wt + (r * 64 +  0 + nl) * 32 + ko);
        v16h b1 = *(const v16h*)(wt + (r * 64 + 16 + nl) * 32 + ko);
        v16h b2 = *(const v16h*)(wt + (r * 64 + 32 + nl) * 32 + ko);
        v16h b3 = *(const v16h*)(wt + (r * 64 + 48 + nl) * 32 + ko);
        acc0 = wmma_f16(a, b0, acc0);
        acc1 = wmma_f16(a, b1, acc1);
        acc2 = wmma_f16(a, b2, acc2);
        acc3 = wmma_f16(a, b3, acc3);
    }

#pragma unroll
    for (int nt = 0; nt < 4; ++nt) {
        const int n = nt * 16 + nl;
        const float inv = g[n] * rsqrtf(var[n] + 1e-5f);
        const float sh  = be[n] - mu[n] * inv;
        const float bb  = bias[n];
        v8f acc = (nt == 0) ? acc0 : (nt == 1) ? acc1 : (nt == 2) ? acc2 : acc3;
#pragma unroll
        for (int rr = 0; rr < 8; ++rr) {
            int mm = mbase + rr + (lane >> 4) * 8;
            int ob = mm / 196;
            int os = mm - ob * 196;
            int oy = os / 14, ox = os - (os / 14) * 14;
            float v = ((float)acc[rr] + bb) * inv + sh;
            v = fmaxf(v, 0.0f);
            act[((ob * 64 + n) * 14 + oy) * 14 + ox] = (_Float16)v;
        }
    }
}

// ---------------------------------------------------------------------------
// K4: 2x2 maxpool on act2 -> flattened h2 [B, 64*7*7] (c-major, matches fc_w)
// ---------------------------------------------------------------------------
__global__ void k_pool2(const _Float16* __restrict__ act, _Float16* __restrict__ h2) {
    int idx = blockIdx.x * 256 + threadIdx.x;
    const int total = BATCH * 64 * 49;
    if (idx >= total) return;
    int px = idx % 7; int t = idx / 7;
    int py = t % 7;   t /= 7;
    int c = t % 64;   int b = t / 64;
    const _Float16* p = act + ((b * 64 + c) * 14 + py * 2) * 14 + px * 2;
    float m = fmaxf(fmaxf((float)p[0], (float)p[1]),
                    fmaxf((float)p[14], (float)p[15]));
    h2[b * 3136 + c * 49 + py * 7 + px] = (_Float16)m;
}

// ---------------------------------------------------------------------------
// K5: FC GEMM via WMMA.  M = 1024, N = 256, K = 3136 (98 steps of 32).
// Epilogue: tanh(acc + fc_b) -> feats f32 [1024,256].
// ---------------------------------------------------------------------------
__global__ void k_fc_wmma(const _Float16* __restrict__ h2,    // [1024,3136]
                          const _Float16* __restrict__ w_h,   // [256,3136]
                          const float* __restrict__ bias,
                          float* __restrict__ feats) {        // [1024,256]
    const int lane = threadIdx.x & 31;
    const int wv   = threadIdx.x >> 5;
    const int mbase = blockIdx.x * 16;
    const int nbase = (blockIdx.y * 4 + wv) * 16;

    const int m = mbase + (lane & 15);
    const int n = nbase + (lane & 15);
    const int ko = (lane >> 4) * 16;

    v8f acc = {};
#pragma unroll 2
    for (int ks = 0; ks < 98; ++ks) {
        const int k = ks * 32 + ko;
        v16h a = *(const v16h*)(h2 + m * 3136 + k);
        v16h b = *(const v16h*)(w_h + n * 3136 + k);
        acc = wmma_f16(a, b, acc);
    }
    const float bb = bias[n];
#pragma unroll
    for (int r = 0; r < 8; ++r) {
        int mm = mbase + r + (lane >> 4) * 8;
        feats[mm * 256 + n] = tanhf((float)acc[r] + bb);
    }
}

// ---------------------------------------------------------------------------
// K6: per-row L2 normalization, f32 -> normalized f16 [1024,256]
// ---------------------------------------------------------------------------
__global__ void k_rownorm(const float* __restrict__ feats, _Float16* __restrict__ out) {
    __shared__ float red[256];
    const int b = blockIdx.x, t = threadIdx.x;
    float v = feats[b * 256 + t];
    red[t] = v * v;
    __syncthreads();
    for (int s = 128; s > 0; s >>= 1) {
        if (t < s) red[t] += red[t + s];
        __syncthreads();
    }
    float inv = rsqrtf(red[0] + 1e-30f);
    out[b * 256 + t] = (_Float16)(v * inv);
}

// ---------------------------------------------------------------------------
// K7: build the full-circuit 256x256 unitary U (complex), single block.
// Circuit: H^{x8}, then 10x [RX layer; S/T phase layer; CZ ladder (diagonal)].
// Emits f32 working copy and f16 copies of Re(U), Im(U) for the WMMA GEMM.
// ---------------------------------------------------------------------------
__global__ void k_build_unitary(const float* __restrict__ qp,
                                float* __restrict__ Ur, float* __restrict__ Ui,
                                _Float16* __restrict__ Urh, _Float16* __restrict__ Uih) {
    const int t = threadIdx.x;
    for (int j = 0; j < 256; ++j) {
        Ur[t * 256 + j] = (t == j) ? 1.0f : 0.0f;
        Ui[t * 256 + j] = 0.0f;
    }
    __syncthreads();

    const float invs = 0.70710678118f;
    // Hadamard on every wire (wire w acts on bit 7-w)
    for (int w = 0; w < 8; ++w) {
        const int mask = 1 << (7 - w);
        for (int p = t; p < 128 * 256; p += 256) {
            int j = p & 255;
            int pr = p >> 8;
            int low = pr & (mask - 1);
            int i0 = ((pr & ~(mask - 1)) << 1) | low;
            int i1 = i0 | mask;
            float ar = Ur[i0 * 256 + j], ai = Ui[i0 * 256 + j];
            float br = Ur[i1 * 256 + j], bi = Ui[i1 * 256 + j];
            Ur[i0 * 256 + j] = (ar + br) * invs; Ui[i0 * 256 + j] = (ai + bi) * invs;
            Ur[i1 * 256 + j] = (ar - br) * invs; Ui[i1 * 256 + j] = (ai - bi) * invs;
        }
        __syncthreads();
    }

    for (int layer = 0; layer < 10; ++layer) {
        // RX(theta) per wire: [[c, -i s],[-i s, c]]
        for (int w = 0; w < 8; ++w) {
            float sn, cs;
            __sincosf(qp[layer * 8 + w] * 0.5f, &sn, &cs);
            const int mask = 1 << (7 - w);
            for (int p = t; p < 128 * 256; p += 256) {
                int j = p & 255;
                int pr = p >> 8;
                int low = pr & (mask - 1);
                int i0 = ((pr & ~(mask - 1)) << 1) | low;
                int i1 = i0 | mask;
                float ar = Ur[i0 * 256 + j], ai = Ui[i0 * 256 + j];
                float br = Ur[i1 * 256 + j], bi = Ui[i1 * 256 + j];
                Ur[i0 * 256 + j] = cs * ar + sn * bi;
                Ui[i0 * 256 + j] = cs * ai - sn * br;
                Ur[i1 * 256 + j] = cs * br + sn * ai;
                Ui[i1 * 256 + j] = cs * bi - sn * ar;
            }
            __syncthreads();
        }
        // Diagonal: S on even wires (bits 0xAA), T on odd wires (bits 0x55),
        // times CZ-ladder sign.
        for (int p = t; p < 256 * 256; p += 256) {
            int i = p >> 8;
            float ang = 1.57079632679f * (float)__popc(i & 0xAA)
                      + 0.78539816340f * (float)__popc(i & 0x55);
            int par = ((i >> 7) & (i >> 6)) ^ ((i >> 5) & (i >> 4)) ^
                      ((i >> 3) & (i >> 2)) ^ ((i >> 1) & i) ^
                      ((i >> 6) & (i >> 5)) ^ ((i >> 4) & (i >> 3)) ^
                      ((i >> 2) & (i >> 1));
            float dr, di;
            __sincosf(ang, &di, &dr);
            if (par & 1) { dr = -dr; di = -di; }
            float vr = Ur[p], vi = Ui[p];
            Ur[p] = vr * dr - vi * di;
            Ui[p] = vr * di + vi * dr;
        }
        __syncthreads();
    }

    for (int p = t; p < 65536; p += 256) {
        Urh[p] = (_Float16)Ur[p];
        Uih[p] = (_Float16)Ui[p];
    }
}

// ---------------------------------------------------------------------------
// K8: quantum GEMM via WMMA: sv = featsn @ U^T (Re & Im share A loads),
// fused |sv|^2 -> probs f32 [1024,256].  M=1024, N=256, K=256.
// ---------------------------------------------------------------------------
__global__ void k_qgemm_wmma(const _Float16* __restrict__ featsn,  // [1024,256]
                             const _Float16* __restrict__ Urh,     // [256,256] (N-major)
                             const _Float16* __restrict__ Uih,
                             float* __restrict__ probs) {          // [1024,256]
    const int lane = threadIdx.x & 31;
    const int wv   = threadIdx.x >> 5;
    const int mbase = blockIdx.x * 16;
    const int nbase = (blockIdx.y * 4 + wv) * 16;

    const int m = mbase + (lane & 15);
    const int n = nbase + (lane & 15);
    const int ko = (lane >> 4) * 16;

    v8f ar = {};
    v8f ai = {};
#pragma unroll
    for (int ks = 0; ks < 8; ++ks) {
        const int k = ks * 32 + ko;
        v16h a  = *(const v16h*)(featsn + m * 256 + k);
        v16h br = *(const v16h*)(Urh + n * 256 + k);
        v16h bi = *(const v16h*)(Uih + n * 256 + k);
        ar = wmma_f16(a, br, ar);
        ai = wmma_f16(a, bi, ai);
    }
#pragma unroll
    for (int r = 0; r < 8; ++r) {
        int mm = mbase + r + (lane >> 4) * 8;
        float re = ar[r], im = ai[r];
        probs[mm * 256 + n] = re * re + im * im;
    }
}

// ---------------------------------------------------------------------------
// K9: q = probs @ ZSIGNS (computed analytically), then MLP 8->128->64->10.
// One block of 128 threads per batch row.
// ---------------------------------------------------------------------------
__global__ void k_head(const float* __restrict__ probs,
                       const float* __restrict__ p1w, const float* __restrict__ p1b,
                       const float* __restrict__ p2w, const float* __restrict__ p2b,
                       const float* __restrict__ p3w, const float* __restrict__ p3b,
                       float* __restrict__ out) {
    __shared__ float q[8];
    __shared__ float z1[128];
    __shared__ float z2[64];
    const int b = blockIdx.x, t = threadIdx.x;
    const float* pr = probs + b * 256;

    if (t < 8) {
        float acc = 0.0f;
        for (int k = 0; k < 256; ++k) {
            float z = 1.0f - 2.0f * (float)((k >> (7 - t)) & 1);
            acc = fmaf(pr[k], z, acc);
        }
        q[t] = acc;
    }
    __syncthreads();

    {
        float acc = p1b[t];
#pragma unroll
        for (int j = 0; j < 8; ++j) acc = fmaf(p1w[t * 8 + j], q[j], acc);
        z1[t] = fmaxf(acc, 0.0f);
    }
    __syncthreads();

    if (t < 64) {
        float acc = p2b[t];
        for (int j = 0; j < 128; ++j) acc = fmaf(p2w[t * 128 + j], z1[j], acc);
        z2[t] = fmaxf(acc, 0.0f);
    }
    __syncthreads();

    if (t < 10) {
        float acc = p3b[t];
        for (int j = 0; j < 64; ++j) acc = fmaf(p3w[t * 64 + j], z2[j], acc);
        out[b * 10 + t] = acc;
    }
}

// ---------------------------------------------------------------------------
extern "C" void kernel_launch(void* const* d_in, const int* in_sizes, int n_in,
                              void* d_out, int out_size, void* d_ws, size_t ws_size,
                              hipStream_t stream) {
    (void)in_sizes; (void)n_in; (void)out_size; (void)ws_size;

    const float* x   = (const float*)d_in[0];
    const float* c1w = (const float*)d_in[1];
    const float* c1b = (const float*)d_in[2];
    const float* g1  = (const float*)d_in[3];
    const float* b1  = (const float*)d_in[4];
    const float* m1  = (const float*)d_in[5];
    const float* v1  = (const float*)d_in[6];
    const float* c2w = (const float*)d_in[7];
    const float* c2b = (const float*)d_in[8];
    const float* g2  = (const float*)d_in[9];
    const float* b2  = (const float*)d_in[10];
    const float* m2  = (const float*)d_in[11];
    const float* v2  = (const float*)d_in[12];
    const float* fcw = (const float*)d_in[13];
    const float* fcb = (const float*)d_in[14];
    const float* qp  = (const float*)d_in[15];
    const float* p1w = (const float*)d_in[16];
    const float* p1b = (const float*)d_in[17];
    const float* p2w = (const float*)d_in[18];
    const float* p2b = (const float*)d_in[19];
    const float* p3w = (const float*)d_in[20];
    const float* p3b = (const float*)d_in[21];
    float* out = (float*)d_out;

    // Workspace carve-up (256B aligned slices)
    char* base = (char*)d_ws;
    size_t off = 0;
    auto take = [&](size_t bytes) -> char* {
        char* p = base + off;
        off = (off + bytes + 255) & ~(size_t)255;
        return p;
    };
    _Float16* pool1   = (_Float16*)take((size_t)BATCH * 16 * 16 * 32 * 2); // 16.8 MB (padded NHWC)
    _Float16* wt2     = (_Float16*)take((size_t)9 * 64 * 32 * 2);
    _Float16* fcw_h   = (_Float16*)take((size_t)256 * 3136 * 2);           // 1.6 MB
    _Float16* act2    = (_Float16*)take((size_t)BATCH * 64 * 14 * 14 * 2); // 25.7 MB
    _Float16* h2      = (_Float16*)take((size_t)BATCH * 3136 * 2);         // 6.4 MB
    float*    feats   = (float*)take((size_t)BATCH * 256 * 4);             // 1 MB
    _Float16* featsn  = (_Float16*)take((size_t)BATCH * 256 * 2);
    float*    Ur      = (float*)take((size_t)256 * 256 * 4);
    float*    Ui      = (float*)take((size_t)256 * 256 * 4);
    _Float16* Urh     = (_Float16*)take((size_t)256 * 256 * 2);
    _Float16* Uih     = (_Float16*)take((size_t)256 * 256 * 2);
    float*    probs   = (float*)take((size_t)BATCH * 256 * 4);             // 1 MB

    // 1) conv1 + bn1 + relu + pool -> padded NHWC pool1 (f16), halo zeroed
    {
        int total = BATCH * 16 * 16 * 32;
        k_conv1_pool_nhwc<<<(total + 255) / 256, 256, 0, stream>>>(
            x, c1w, c1b, g1, b1, m1, v1, pool1);
    }
    // 2) weight repacks/conversions for WMMA
    k_repack_w2<<<(9 * 64 * 32 + 255) / 256, 256, 0, stream>>>(c2w, wt2);
    k_f32_to_f16<<<(256 * 3136 + 255) / 256, 256, 0, stream>>>(fcw, fcw_h, 256 * 3136);

    // 3) conv2 = 9 shifted WMMA GEMMs + bias + bn2 + relu -> act2 (f16)
    //    12544 M-tiles, 4 waves/block, each wave does full N=64
    k_conv2_wmma<<<3136, 128, 0, stream>>>(pool1, wt2, c2b, g2, b2, m2, v2, act2);

    // 4) 2x2 maxpool -> h2 flattened (f16)
    {
        int total = BATCH * 64 * 49;
        k_pool2<<<(total + 255) / 256, 256, 0, stream>>>(act2, h2);
    }

    // 5) FC GEMM (WMMA) + tanh -> feats (f32)
    k_fc_wmma<<<dim3(64, 4), 128, 0, stream>>>(h2, fcw_h, fcb, feats);

    // 6) row-wise L2 normalize -> featsn (f16)
    k_rownorm<<<BATCH, 256, 0, stream>>>(feats, featsn);

    // 7) build full-circuit unitary (one block), emit f16 Re/Im
    k_build_unitary<<<1, 256, 0, stream>>>(qp, Ur, Ui, Urh, Uih);

    // 8) quantum GEMMs (WMMA, shared A loads) + |sv|^2 -> probs (f32)
    k_qgemm_wmma<<<dim3(64, 4), 128, 0, stream>>>(featsn, Urh, Uih, probs);

    // 9) Z-expectations + MLP head -> out [1024,10]
    k_head<<<BATCH, 128, 0, stream>>>(probs, p1w, p1b, p2w, p2b, p3w, p3b, out);
}
